// MultiHeadAttentionRoPE_49460843381665
// MI455X (gfx1250) — matrix operand order repaired and checked by reading, not compiled, not run on hardware
//
#include <hip/hip_runtime.h>
#include <hip/hip_bf16.h>

typedef __bf16 bf16;
typedef __attribute__((ext_vector_type(16))) __bf16 v16bf;
typedef __attribute__((ext_vector_type(8)))  __bf16 v8bf;
typedef __attribute__((ext_vector_type(8)))  float  v8f;

static constexpr int Bv = 4, Sv = 2048, Dv = 1024, Hv = 16, DKv = 64, DVv = 64;

#define WMMA_BF16(A, B, C) \
    __builtin_amdgcn_wmma_f32_16x16x32_bf16(false, (A), false, (B), (short)0, (C), false, false)

// ---- async global->LDS copy (gfx1250 ASYNCcnt path), with safe fallback ----
#if __has_builtin(__builtin_amdgcn_global_load_async_to_lds_b128)
#define ASYNC_LDS 1
typedef int v4i __attribute__((__vector_size__(4 * sizeof(int))));
typedef __attribute__((address_space(1))) v4i g_v4i;
typedef __attribute__((address_space(3))) v4i l_v4i;
__device__ __forceinline__ void async_cp16(const void* g, void* l) {
    __builtin_amdgcn_global_load_async_to_lds_b128((g_v4i*)g, (l_v4i*)l, 0, 0);
}
#define WAIT_ASYNC(n) asm volatile("s_wait_asynccnt " #n ::: "memory")
#else
#define ASYNC_LDS 0
__device__ __forceinline__ void async_cp16(const void* g, void* l) {
    *(float4*)l = *(const float4*)g;   // falls back to global_load + ds_store
}
#define WAIT_ASYNC(n)
#endif

// pack 8 floats (two float4) into v16bf slots [o..o+7]
__device__ __forceinline__ void pack8(v16bf& d, int o, float4 a, float4 b) {
    d[o + 0] = (bf16)a.x; d[o + 1] = (bf16)a.y; d[o + 2] = (bf16)a.z; d[o + 3] = (bf16)a.w;
    d[o + 4] = (bf16)b.x; d[o + 5] = (bf16)b.y; d[o + 6] = (bf16)b.z; d[o + 7] = (bf16)b.w;
}

__device__ __forceinline__ v16bf cat8(v8bf a, v8bf b) {
    v16bf r;
#pragma unroll
    for (int i = 0; i < 8; ++i) { r[i] = a[i]; r[i + 8] = b[i]; }
    return r;
}

// ---------------------------------------------------------------------------
// GEMM: C[M,N] = A[M,K] * W[N,K]^T  (fp32 in, bf16 WMMA core, f32 accum)
// Block 256 threads (8 waves); tile 128(M) x 64(N); K-step 32.
// A/W tiles staged in LDS via async loads, double-buffered.
// MODE 0: RoPE epilogue, bf16 [B,H,S,64]; MODE 1: bf16 [B,H,64,S]; MODE 2: f32.
// ---------------------------------------------------------------------------
static constexpr int LSTR = 36;  // LDS row stride in dwords (32 data + 4 pad)

template <int MODE>
__global__ __launch_bounds__(256) void gemm_bf16_wmma(
    const float* __restrict__ A, const float* __restrict__ W,
    const float* __restrict__ rope_cos, const float* __restrict__ rope_sin,
    bf16* __restrict__ out_bf, float* __restrict__ out_f32,
    int M, int N, int K)
{
    __shared__ float lds_a[2][128 * LSTR];   // 2 x 18 KB
    __shared__ float lds_w[2][64 * LSTR];    // 2 x  9 KB

    const int tid  = threadIdx.x;
    const int lane = tid & 31;
    const int wave = tid >> 5;
    const int g    = lane >> 4;
    const int ln   = lane & 15;
    const int n0   = blockIdx.x * 64;
    const int m_b  = blockIdx.y * 128;

    const float* Ab = A + (long)m_b * K;
    const float* Wb = W + (long)n0 * K;

    // stage one K-tile (k0) into buffer `buf`: 6 async b128 per thread
    auto stage = [&](int k0, int buf) {
#pragma unroll
        for (int i = 0; i < 4; ++i) {               // A: 128 rows x 8 chunks
            const int c = tid + i * 256;
            const int row = c >> 3, qc = c & 7;
            async_cp16(Ab + (long)row * K + k0 + qc * 4,
                       &lds_a[buf][row * LSTR + qc * 4]);
        }
#pragma unroll
        for (int i = 0; i < 2; ++i) {               // W: 64 rows x 8 chunks
            const int c = tid + i * 256;
            const int row = c >> 3, qc = c & 7;
            async_cp16(Wb + (long)row * K + k0 + qc * 4,
                       &lds_w[buf][row * LSTR + qc * 4]);
        }
    };

    v8f acc[4] = {};
    stage(0, 0);

    for (int k0 = 0; k0 < K; k0 += 32) {
        const int cur = (k0 >> 5) & 1;
        if (k0 + 32 < K) {
            stage(k0 + 32, cur ^ 1);
            WAIT_ASYNC(6);      // current tile's 6 copies done; prefetch in flight
        } else {
            WAIT_ASYNC(0);
        }
        __syncthreads();

        const float* la = lds_a[cur];
        const float* lw = lds_w[cur];

        v16bf af;
        {
            const float* ar = la + (wave * 16 + ln) * LSTR;
            const float4* pl = (const float4*)(ar + g * 8);
            const float4* ph = (const float4*)(ar + 16 + g * 8);
            pack8(af, 0, pl[0], pl[1]);
            pack8(af, 8, ph[0], ph[1]);
        }
#pragma unroll
        for (int t = 0; t < 4; ++t) {
            const float4* pw = (const float4*)(lw + (t * 16 + ln) * LSTR + g * 8);
            v16bf bfr;
            pack8(bfr, 0, pw[0], pw[1]);
            pack8(bfr, 8, pw[4], pw[5]);
            acc[t] = WMMA_BF16(af, bfr, acc[t]);
        }
        __syncthreads();
    }

    const int h = n0 / 64;  // head index for MODE 0/1 (N==H*64, BN==64)
    const int m_w = m_b + wave * 16;
#pragma unroll
    for (int v = 0; v < 8; ++v) {
        const int m = m_w + v + 8 * g;   // C layout: row = vgpr + 8*halfwave
        if constexpr (MODE == 0) {
            const int b = m / Sv, s = m % Sv;
            const long base = ((long)(b * Hv + h) * Sv + s) * 64;
#pragma unroll
            for (int t = 0; t < 2; ++t) {
                const int dlo = t * 16 + ln;
                const int dhi = dlo + 32;
                const float clo = rope_cos[s * 64 + dlo], slo = rope_sin[s * 64 + dlo];
                const float chi = rope_cos[s * 64 + dhi], shi = rope_sin[s * 64 + dhi];
                const float xlo = acc[t][v], xhi = acc[t + 2][v];
                out_bf[base + dlo] = (bf16)(xlo * clo - xhi * slo);
                out_bf[base + dhi] = (bf16)(xhi * chi + xlo * shi);
            }
        } else if constexpr (MODE == 1) {
            const int b = m / Sv, s = m % Sv;
#pragma unroll
            for (int t = 0; t < 4; ++t) {
                const int d = t * 16 + ln;
                out_bf[((long)(b * Hv + h) * 64 + d) * Sv + s] = (bf16)acc[t][v];
            }
        } else {
#pragma unroll
            for (int t = 0; t < 4; ++t)
                out_f32[(long)m * N + n0 + t * 16 + ln] = acc[t][v];
        }
    }
}

// ---------------------------------------------------------------------------
// Flash attention (non-causal, online softmax), bf16 WMMA.
// Block 128 threads (4 waves); each wave owns 16 query rows.
// Shared K (32x64 bf16) and V^T (64x32 bf16) tiles staged via async loads,
// double-buffered.  S^T = K x Q^T so score C-layout == P's A-frag layout.
// ---------------------------------------------------------------------------
static constexpr int KSTR = 144;  // bytes/row: 128 data + 16 pad
static constexpr int VSTR = 80;   // bytes/row:  64 data + 16 pad

__global__ __launch_bounds__(128) void flash_attn_wmma(
    const bf16* __restrict__ qh,   // [B,H,S,64]
    const bf16* __restrict__ kh,   // [B,H,S,64]
    const bf16* __restrict__ vt,   // [B,H,64,S]
    float* __restrict__ attn_out)  // [B,S,H*64]
{
    __shared__ __align__(16) char lds_k[2][32 * KSTR];
    __shared__ __align__(16) char lds_v[2][64 * VSTR];

    const int tid  = threadIdx.x;
    const int lane = tid & 31;
    const int wave = tid >> 5;
    const int g  = lane >> 4;
    const int ln = lane & 15;
    const int bh = blockIdx.y;
    const int b  = bh / Hv, h = bh % Hv;
    const int q0 = blockIdx.x * 64 + wave * 16;

    const bf16* qbase = qh + (long)bh * Sv * 64;
    const bf16* kbase = kh + (long)bh * Sv * 64;
    const bf16* vbase = vt + (long)bh * 64 * Sv;

    // stage K rows [k0,k0+32) and V^T cols [k0,k0+32): 4 async b128 per thread
    auto stage = [&](int k0, int buf) {
#pragma unroll
        for (int i = 0; i < 2; ++i) {               // K: 32 rows x 8 chunks
            const int c = tid + i * 128;
            const int row = c >> 3, qc = c & 7;
            async_cp16(kbase + (long)(k0 + row) * 64 + qc * 8,
                       &lds_k[buf][row * KSTR + qc * 16]);
        }
#pragma unroll
        for (int i = 0; i < 2; ++i) {               // V^T: 64 rows x 4 chunks
            const int c = tid + i * 128;
            const int row = c >> 2, qc = c & 3;
            async_cp16(vbase + (long)row * Sv + k0 + qc * 8,
                       &lds_v[buf][row * VSTR + qc * 16]);
        }
    };

    // Q as B-fragment: lane=q col; chunk c: elems e -> d = 32c + 16g + e
    v16bf qf[2];
#pragma unroll
    for (int c = 0; c < 2; ++c)
        qf[c] = *(const v16bf*)(qbase + (long)(q0 + ln) * 64 + c * 32 + g * 16);

    v8f acc[4] = {};
    float m_i = -1e30f, l_i = 0.f;
    const float scale = 0.125f;  // DK^-0.5

    stage(0, 0);

    for (int k0 = 0; k0 < Sv; k0 += 32) {
        const int cur = (k0 >> 5) & 1;
        if (k0 + 32 < Sv) {
            stage(k0 + 32, cur ^ 1);
            WAIT_ASYNC(4);
        } else {
            WAIT_ASYNC(0);
        }
        __syncthreads();

        const char* lk = lds_k[cur];
        const char* lv = lds_v[cur];

        // scores S^T tiles: st[kt][v] = s[k = k0+16kt+(v+8g)][q = q0+ln]
        v8f st[2];
#pragma unroll
        for (int kt = 0; kt < 2; ++kt) {
            const char* kr = lk + (kt * 16 + ln) * KSTR;
            v16bf ka0 = cat8(*(const v8bf*)(kr + 32 * g),      *(const v8bf*)(kr + 16 + 32 * g));
            v16bf ka1 = cat8(*(const v8bf*)(kr + 64 + 32 * g), *(const v8bf*)(kr + 80 + 32 * g));
            v8f s = {};
            s = WMMA_BF16(ka0, qf[0], s);
            s = WMMA_BF16(ka1, qf[1], s);
            st[kt] = s;
        }
        float rowmax = -1e30f;
#pragma unroll
        for (int kt = 0; kt < 2; ++kt)
#pragma unroll
            for (int v = 0; v < 8; ++v) {
                st[kt][v] *= scale;
                rowmax = fmaxf(rowmax, st[kt][v]);
            }
        rowmax = fmaxf(rowmax, __shfl_xor(rowmax, 16, 32));
        const float m_new  = fmaxf(m_i, rowmax);
        const float rscale = __expf(m_i - m_new);

        v16bf pa;
        float psum = 0.f;
#pragma unroll
        for (int v = 0; v < 8; ++v) {
            const float p0 = __expf(st[0][v] - m_new);
            const float p1 = __expf(st[1][v] - m_new);
            psum += p0 + p1;
            pa[v]     = (bf16)p0;   // A-frag elems 0..7  <-> kv = 8g+{0..7}
            pa[v + 8] = (bf16)p1;   // elems 8..15        <-> kv = 16+8g+{0..7}
        }
        psum += __shfl_xor(psum, 16, 32);
        l_i = l_i * rscale + psum;
        m_i = m_new;

#pragma unroll
        for (int v = 0; v < 8; ++v) {
            const float rr = __shfl(rscale, v + 8 * g, 32);
#pragma unroll
            for (int t = 0; t < 4; ++t) acc[t][v] *= rr;
        }
#pragma unroll
        for (int t = 0; t < 4; ++t) {
            const char* vr = lv + (t * 16 + ln) * VSTR + 16 * g;
            v16bf vf = cat8(*(const v8bf*)(vr), *(const v8bf*)(vr + 32));
            acc[t] = WMMA_BF16(pa, vf, acc[t]);
        }
        __syncthreads();
    }

#pragma unroll
    for (int v = 0; v < 8; ++v) {
        const float lr  = __shfl(l_i, v + 8 * g, 32);
        const float inv = 1.0f / lr;
        const int s = q0 + v + 8 * g;
        float* orow = attn_out + ((long)b * Sv + s) * (Hv * DVv) + h * 64;
#pragma unroll
        for (int t = 0; t < 4; ++t) orow[t * 16 + ln] = acc[t][v] * inv;
    }
}

extern "C" void kernel_launch(void* const* d_in, const int* in_sizes, int n_in,
                              void* d_out, int out_size, void* d_ws, size_t ws_size,
                              hipStream_t stream) {
    const float* q    = (const float*)d_in[0];
    const float* k    = (const float*)d_in[1];
    const float* v    = (const float*)d_in[2];
    const float* rc   = (const float*)d_in[3];
    const float* rs   = (const float*)d_in[4];
    const float* w_qs = (const float*)d_in[5];
    const float* w_ks = (const float*)d_in[6];
    const float* w_vs = (const float*)d_in[7];
    const float* w_fc = (const float*)d_in[8];
    float* out = (float*)d_out;

    char* ws = (char*)d_ws;
    bf16*  q_bf    = (bf16*)(ws);                          // 16 MB
    bf16*  k_bf    = (bf16*)(ws + ((size_t)16 << 20));     // 16 MB
    bf16*  vt_bf   = (bf16*)(ws + ((size_t)32 << 20));     // 16 MB
    float* attn_ws = (float*)(ws + ((size_t)48 << 20));    // 32 MB

    const int M = Bv * Sv;   // 8192
    const int N = Dv;        // 1024
    const int K = Dv;        // 1024

    dim3 gblk(256);
    dim3 ggrid(N / 64, M / 128);
    gemm_bf16_wmma<0><<<ggrid, gblk, 0, stream>>>(q, w_qs, rc, rs, q_bf,  nullptr, M, N, K);
    gemm_bf16_wmma<0><<<ggrid, gblk, 0, stream>>>(k, w_ks, rc, rs, k_bf,  nullptr, M, N, K);
    gemm_bf16_wmma<1><<<ggrid, gblk, 0, stream>>>(v, w_vs, nullptr, nullptr, vt_bf, nullptr, M, N, K);

    flash_attn_wmma<<<dim3(Sv / 64, Bv * Hv), dim3(128), 0, stream>>>(q_bf, k_bf, vt_bf, attn_ws);

    gemm_bf16_wmma<2><<<ggrid, gblk, 0, stream>>>(attn_ws, w_fc, nullptr, nullptr, nullptr, out, M, N, K);
}
